// CrossAttention_90400471646744
// MI455X (gfx1250) — matrix operand order repaired and checked by reading, not compiled, not run on hardware
//
#include <hip/hip_runtime.h>
#include <hip/hip_bf16.h>
#include <stdint.h>

// ---------------------------------------------------------------------------
// CrossAttention with per-query dynamic value projection, fused for MI455X.
//   out = ( Σ_d wctx[h,m,d] * Σ_c x[m,c] * Wv[c, d*512 + e] ) @ Wo
// All GEMMs: v_wmma_f32_16x16x32_bf16 (wave32, fp32 accum).
// Wv (536 MB fp32) streamed from HBM exactly once via double-buffered
// global_load_async_to_lds_b128 (ASYNCcnt), converted to bf16 in LDS.
// ---------------------------------------------------------------------------

typedef __attribute__((ext_vector_type(16))) __bf16 v16bf;
typedef __attribute__((ext_vector_type(8)))  __bf16 v8bf;
typedef __attribute__((ext_vector_type(8)))  float  v8f;
typedef __attribute__((ext_vector_type(4)))  float  f32x4;

#define B_   8
#define NQ_  77
#define NK_  128
#define D_   512
#define H_   8
#define DH_  64
#define MQ   616       // B*NQ real rows
#define MQP  624       // padded to 39 m-tiles of 16
#define MT_Q 39
#define MK   1024      // B*NK rows
#define NSLAB 32
#define DSLAB 16       // d's per workgroup in heavy kernel

// ---- WMMA + fragment helpers ---------------------------------------------

__device__ __forceinline__ v8f wmma_bf16(v16bf a, v16bf b, v8f c) {
  return __builtin_amdgcn_wmma_f32_16x16x32_bf16(false, a, false, b,
                                                 (short)0, c, false, false);
}

__device__ __forceinline__ v16bf pack16(v8bf lo, v8bf hi) {
  v16bf r;
#pragma unroll
  for (int i = 0; i < 8; ++i) { r[i] = lo[i]; r[i + 8] = hi[i]; }
  return r;
}

// A fragment (16x32) from row-major bf16 [*, lda]; 16B-aligned rows required.
// ISA layout: lanes 0-15 row=lane, K={0..7,16..23}; lanes 16-31 row=lane-16,
// K={8..15,24..31}.
__device__ __forceinline__ v16bf afrag_rm(const __bf16* A, int lda, int m0, int k0) {
  int l = threadIdx.x & 31;
  const __bf16* p = A + (size_t)(m0 + (l & 15)) * lda + k0 + ((l < 16) ? 0 : 8);
  return pack16(*(const v8bf*)p, *(const v8bf*)(p + 16));
}

// Same, converting from fp32 source (vector loads).
__device__ __forceinline__ v16bf afrag_rm_f32(const float* A, int lda, int m0, int k0) {
  int l = threadIdx.x & 31;
  const float* p = A + (size_t)(m0 + (l & 15)) * lda + k0 + ((l < 16) ? 0 : 8);
  f32x4 a0 = *(const f32x4*)p,        a1 = *(const f32x4*)(p + 4);
  f32x4 b0 = *(const f32x4*)(p + 16), b1 = *(const f32x4*)(p + 20);
  v16bf r;
#pragma unroll
  for (int i = 0; i < 4; ++i) {
    r[i] = (__bf16)a0[i]; r[i + 4] = (__bf16)a1[i];
    r[i + 8] = (__bf16)b0[i]; r[i + 12] = (__bf16)b1[i];
  }
  return r;
}

// B fragment (32x16) from K-major [N][K] bf16 source (contiguous per lane).
// Layout: lanes 0-15 col=lane, K={0..7,16..23}; lanes 16-31 col=lane-16, K={8..15,24..31}.
__device__ __forceinline__ v16bf bfrag_km(const __bf16* Bm, int ldk, int k0, int n0) {
  int l = threadIdx.x & 31;
  const __bf16* p = Bm + (size_t)(n0 + (l & 15)) * ldk + k0 + ((l < 16) ? 0 : 8);
  return pack16(*(const v8bf*)p, *(const v8bf*)(p + 16));
}

// ---- async global -> LDS staging (CDNA5, ASYNCcnt) ------------------------
// One 32B chunk per thread; instruction offset applies to BOTH addresses.

__device__ __forceinline__ void async_tile_load32(const float* gsrc, uint32_t ldsoff) {
  asm volatile("global_load_async_to_lds_b128 %0, %1, off"
               :: "v"(ldsoff), "v"(gsrc) : "memory");
  asm volatile("global_load_async_to_lds_b128 %0, %1, off offset:16"
               :: "v"(ldsoff), "v"(gsrc) : "memory");
}

__device__ __forceinline__ void wait_async0() {
  asm volatile("s_wait_asynccnt 0x0" ::: "memory");
}

// ---- small utility kernels ------------------------------------------------

__global__ void zero_u32_kernel(uint32_t* p, int n) {
  int i = blockIdx.x * blockDim.x + threadIdx.x;
  if (i < n) p[i] = 0u;
}

__global__ void cvt_f32_bf16_kernel(const float* src, __bf16* dst, int n) {
  int i = blockIdx.x * blockDim.x + threadIdx.x;
  if (i < n) dst[i] = (__bf16)src[i];
}

// Transpose + convert: dst[c*rows + r] = (bf16)src[r*cols + c], batched over z.
__global__ __launch_bounds__(256)
void transpose_cvt_kernel(const float* src, __bf16* dst, int rows, int cols) {
  __shared__ float tile[32][33];
  src += (size_t)blockIdx.z * rows * cols;
  dst += (size_t)blockIdx.z * rows * cols;
  int tx = threadIdx.x & 31, ty = threadIdx.x >> 5;
#pragma unroll
  for (int i = ty; i < 32; i += 8)
    tile[i][tx] = src[(size_t)(blockIdx.y * 32 + i) * cols + blockIdx.x * 32 + tx];
  __syncthreads();
#pragma unroll
  for (int i = ty; i < 32; i += 8)
    dst[(size_t)(blockIdx.x * 32 + i) * rows + blockIdx.y * 32 + tx] =
        (__bf16)tile[tx][i];
}

// ---- K1: Q / K projections (bf16 WMMA), writes [b][h][seq][64] ------------

__global__ __launch_bounds__(256)
void proj_kernel(const __bf16* A, const __bf16* WT, __bf16* outb,
                 int Mreal, int SEQreal, int SEQpad, float scale) {
  int w = threadIdx.x >> 5, l = threadIdx.x & 31;
  int mt = blockIdx.x;
  int nt = blockIdx.y * 8 + w;          // 32 n-tiles over N=512
  v8f acc = {};
#pragma unroll 4
  for (int kc = 0; kc < 16; ++kc) {
    v16bf a = afrag_rm(A, D_, mt * 16, kc * 32);
    v16bf b = bfrag_km(WT, D_, kc * 32, nt * 16);
    acc = wmma_bf16(a, b, acc);
  }
  int hi = (l >> 4) << 3;
#pragma unroll
  for (int j = 0; j < 8; ++j) {
    int row = mt * 16 + j + hi;
    if (row < Mreal) {
      int col = nt * 16 + (l & 15);
      int b = row / SEQreal, s = row % SEQreal;
      int hh = col >> 6, dh = col & 63;
      outb[(((size_t)(b * H_ + hh) * SEQpad) + s) * DH_ + dh] = (__bf16)(acc[j] * scale);
    }
  }
}

// ---- K2: sim = QK^T, softmax, wctx = attn @ context  (one WG per (b,h)) ---

__global__ __launch_bounds__(256)
void attn_wctx_kernel(const __bf16* Qb, const __bf16* Kb, const __bf16* ctxT,
                      float* wctx) {
  int b = blockIdx.x, h = blockIdx.y;
  int tid = threadIdx.x, w = tid >> 5, l = tid & 31;
  __shared__ float  simS[80 * 129];
  __shared__ __bf16 attnS[80 * 128];
  const __bf16* Q = Qb + (size_t)(b * H_ + h) * 80 * DH_;
  const __bf16* K = Kb + (size_t)(b * H_ + h) * NK_ * DH_;

  // sim [80 x 128] : 5 m-tiles x 8 n-tiles, K = 64 (2 chunks)
#pragma unroll
  for (int t = 0; t < 5; ++t) {
    int tile = w * 5 + t;
    int mt = tile >> 3, nt = tile & 7;
    v8f acc = {};
#pragma unroll
    for (int kc = 0; kc < 2; ++kc) {
      v16bf a = afrag_rm(Q, DH_, mt * 16, kc * 32);
      v16bf bb = bfrag_km(K, DH_, kc * 32, nt * 16);  // K^T: n=key, k=dh
      acc = wmma_bf16(a, bb, acc);
    }
    int hi = (l >> 4) << 3;
#pragma unroll
    for (int j = 0; j < 8; ++j)
      simS[(mt * 16 + j + hi) * 129 + nt * 16 + (l & 15)] = acc[j];
  }
  __syncthreads();

  // softmax per row (fp32), write bf16 attn
  if (tid < 80) {
    float mx = -3.0e38f;
    for (int k = 0; k < NK_; ++k) mx = fmaxf(mx, simS[tid * 129 + k]);
    float sm = 0.f;
    for (int k = 0; k < NK_; ++k) {
      float e = __expf(simS[tid * 129 + k] - mx);
      simS[tid * 129 + k] = e;
      sm += e;
    }
    float inv = 1.f / sm;
    for (int k = 0; k < NK_; ++k)
      attnS[tid * 128 + k] = (__bf16)(simS[tid * 129 + k] * inv);
  }
  __syncthreads();

  // wctx[h, b*77+q, :] = attn [80x128] @ context[b] [128x512] (B from ctxT)
  const __bf16* CT = ctxT + (size_t)b * D_ * NK_;   // [d(512)][k(128)]
  v8f acc[5][4] = {};
#pragma unroll
  for (int kc = 0; kc < 4; ++kc) {
    v16bf bf[4];
#pragma unroll
    for (int ni = 0; ni < 4; ++ni)
      bf[ni] = bfrag_km(CT, NK_, kc * 32, (ni * 8 + w) * 16);
#pragma unroll
    for (int mt = 0; mt < 5; ++mt) {
      v16bf a = afrag_rm(attnS, 128, mt * 16, kc * 32);
#pragma unroll
      for (int ni = 0; ni < 4; ++ni) acc[mt][ni] = wmma_bf16(a, bf[ni], acc[mt][ni]);
    }
  }
  int hi = (l >> 4) << 3;
#pragma unroll
  for (int mt = 0; mt < 5; ++mt)
#pragma unroll
    for (int ni = 0; ni < 4; ++ni)
#pragma unroll
      for (int j = 0; j < 8; ++j) {
        int q = mt * 16 + j + hi;
        if (q < NQ_) {
          int n = (ni * 8 + w) * 16 + (l & 15);
          wctx[((size_t)h * MQP + b * NQ_ + q) * D_ + n] = acc[mt][ni][j];
        }
      }
  // zero the 8 padded wctx rows (616..623) once per h
  if (b == 0)
    for (int i = tid; i < 8 * D_; i += 256)
      wctx[((size_t)h * MQP + MQ) * D_ + i] = 0.f;
}

// ---- K3: heavy fused contraction ------------------------------------------
// out_pre[m, h*64+n] += sum_{d in slab} wctx[h,m,d] * sum_c x[m,c]*Wv[c,d*512+h*64+n]
// grid = (NSLAB, H). All 624 rows resident (39 m-tiles over 8 waves).
// Wv tiles [32c x 64n] fp32 double-buffered via async LDS loads.

__global__ __launch_bounds__(256)
void heavy_kernel(const float* __restrict__ Wv, const __bf16* __restrict__ xbf,
                  const float* __restrict__ wctx, float* __restrict__ outp) {
  int slab = blockIdx.x, h = blockIdx.y;
  int tid = threadIdx.x, w = tid >> 5, l = tid & 31;
  extern __shared__ char smem[];                 // dynamic => LDS offset 0
  float*  stage = (float*)smem;                  // 2 x [32][64] fp32 (16 KB)
  __bf16* Bs    = (__bf16*)(smem + 16384);       // [64][40] bf16, 16B rows

  v8f acc[5][4] = {};
  int nmt = (w < 7) ? 5 : 4;                     // m-tiles: mt = w + 8*i

  // async staging map: thread -> 8 consecutive floats of row r
  int r = tid >> 3, n0g = (tid & 7) * 8;
  const float* gbase = Wv + (size_t)r * (D_ * D_)
                          + (size_t)(slab * DSLAB) * D_ + h * DH_ + n0g;
  uint32_t soff = (uint32_t)((r * 64 + n0g) * 4);

  async_tile_load32(gbase, soff);                // tile (cc=0, dd=0) -> buf 0
  int buf = 0;
  v16bf xf[5];

  for (int it = 0; it < 256; ++it) {
    int cc = it >> 4, dd = it & 15;
    if (dd == 0) {
#pragma unroll
      for (int i = 0; i < 5; ++i)
        if (i < nmt) xf[i] = afrag_rm(xbf, D_, (w + i * 8) * 16, cc * 32);
    }

    wait_async0();                               // stage[buf] resident
    __syncthreads();
    if (it < 255) {                              // prefetch next tile
      int it2 = it + 1;
      const float* gsrc = gbase + (size_t)(it2 >> 4) * 32 * (D_ * D_)
                                + (size_t)(it2 & 15) * D_;
      async_tile_load32(gsrc, (uint32_t)((buf ^ 1) * 8192) + soff);
    }
    {
      // convert stage[buf] fp32 [c][n] -> Bs bf16 [n][40] (K-major, aligned)
      const float* st = stage + buf * 2048;
      int n = tid >> 2, r0 = (tid & 3) * 8;
      v8bf vb;
#pragma unroll
      for (int j = 0; j < 8; ++j) vb[j] = (__bf16)st[(r0 + j) * 64 + n];
      *(v8bf*)(Bs + n * 40 + r0) = vb;
    }
    __syncthreads();

    v16bf bf[4];
#pragma unroll
    for (int ni = 0; ni < 4; ++ni) bf[ni] = bfrag_km(Bs, 40, 0, ni * 16);

    int dg = slab * DSLAB + dd;
    for (int i = 0; i < nmt; ++i) {
      int m = (w + i * 8) * 16 + (l & 15);
      float s = wctx[((size_t)h * MQP + m) * D_ + dg];
      v16bf a;
#pragma unroll
      for (int e = 0; e < 16; ++e) a[e] = (__bf16)(s * (float)xf[i][e]);
#pragma unroll
      for (int ni = 0; ni < 4; ++ni) acc[i][ni] = wmma_bf16(a, bf[ni], acc[i][ni]);
    }
    buf ^= 1;
  }

  int hi = (l >> 4) << 3;
  for (int i = 0; i < nmt; ++i) {
    int mt = w + i * 8;
#pragma unroll
    for (int ni = 0; ni < 4; ++ni)
#pragma unroll
      for (int j = 0; j < 8; ++j) {
        int m = mt * 16 + j + hi;
        if (m < MQ)
          atomicAdd(&outp[(size_t)m * D_ + h * DH_ + ni * 16 + (l & 15)], acc[i][ni][j]);
      }
  }
}

// ---- K4: final projection out_pre @ Wo -> d_out ---------------------------

__global__ __launch_bounds__(256)
void final_kernel(const float* opre, const __bf16* WoT, float* out) {
  int w = threadIdx.x >> 5, l = threadIdx.x & 31;
  int mt = blockIdx.x;
  int nt = blockIdx.y * 8 + w;
  v8f acc = {};
#pragma unroll 4
  for (int kc = 0; kc < 16; ++kc) {
    v16bf a = afrag_rm_f32(opre, D_, mt * 16, kc * 32);
    v16bf b = bfrag_km(WoT, D_, kc * 32, nt * 16);
    acc = wmma_bf16(a, b, acc);
  }
  int hi = (l >> 4) << 3;
#pragma unroll
  for (int j = 0; j < 8; ++j) {
    int row = mt * 16 + j + hi;
    if (row < MQ) out[(size_t)row * D_ + nt * 16 + (l & 15)] = acc[j];
  }
}

// ---- host side ------------------------------------------------------------

static constexpr size_t OFF_XBF  = 0;                                   // [624][512] bf16
static constexpr size_t OFF_CTXB = OFF_XBF  + (size_t)MQP * D_ * 2;     // [1024][512] bf16
static constexpr size_t OFF_CTXT = OFF_CTXB + (size_t)B_ * NK_ * D_ * 2;// [8][512][128] bf16
static constexpr size_t OFF_WQT  = OFF_CTXT + (size_t)B_ * NK_ * D_ * 2;// [512][512] bf16 (K-major)
static constexpr size_t OFF_WKT  = OFF_WQT  + (size_t)D_ * D_ * 2;
static constexpr size_t OFF_WOT  = OFF_WKT  + (size_t)D_ * D_ * 2;
static constexpr size_t OFF_QB   = OFF_WOT  + (size_t)D_ * D_ * 2;      // [8][8][80][64]
static constexpr size_t OFF_KB   = OFF_QB   + (size_t)B_ * H_ * 80 * DH_ * 2;
static constexpr size_t OFF_WCTX = OFF_KB   + (size_t)B_ * H_ * NK_ * DH_ * 2;
static constexpr size_t OFF_OPRE = OFF_WCTX + (size_t)H_ * MQP * D_ * 4;

extern "C" void kernel_launch(void* const* d_in, const int* in_sizes, int n_in,
                              void* d_out, int out_size, void* d_ws, size_t ws_size,
                              hipStream_t stream) {
  const float* x   = (const float*)d_in[0];
  const float* ctx = (const float*)d_in[1];
  const float* Wq  = (const float*)d_in[2];
  const float* Wk  = (const float*)d_in[3];
  const float* Wv  = (const float*)d_in[4];
  const float* Wo  = (const float*)d_in[5];
  float* out = (float*)d_out;
  char* ws = (char*)d_ws;

  __bf16* xbf  = (__bf16*)(ws + OFF_XBF);
  __bf16* ctxb = (__bf16*)(ws + OFF_CTXB);
  __bf16* ctxT = (__bf16*)(ws + OFF_CTXT);
  __bf16* wqT  = (__bf16*)(ws + OFF_WQT);
  __bf16* wkT  = (__bf16*)(ws + OFF_WKT);
  __bf16* woT  = (__bf16*)(ws + OFF_WOT);
  __bf16* qb   = (__bf16*)(ws + OFF_QB);
  __bf16* kb   = (__bf16*)(ws + OFF_KB);
  float*  wctx = (float*)(ws + OFF_WCTX);
  float*  opre = (float*)(ws + OFF_OPRE);

  const float scale = 0.3535533905932738f;   // 64^-0.25
  auto blks = [](int n) { return (n + 255) / 256; };

  // zero padded / accumulated buffers
  zero_u32_kernel<<<blks(MQP * D_ / 2), 256, 0, stream>>>((uint32_t*)xbf, MQP * D_ / 2);
  zero_u32_kernel<<<blks(B_ * H_ * 80 * DH_ / 2), 256, 0, stream>>>((uint32_t*)qb, B_ * H_ * 80 * DH_ / 2);
  zero_u32_kernel<<<blks(MQP * D_), 256, 0, stream>>>((uint32_t*)opre, MQP * D_);

  // conversions (row-major) + transpose-conversions (K-major B operands)
  cvt_f32_bf16_kernel<<<blks(MQ * D_), 256, 0, stream>>>(x, xbf, MQ * D_);
  cvt_f32_bf16_kernel<<<blks(B_ * NK_ * D_), 256, 0, stream>>>(ctx, ctxb, B_ * NK_ * D_);
  transpose_cvt_kernel<<<dim3(16, 4, 8), 256, 0, stream>>>(ctx, ctxT, NK_, D_);
  transpose_cvt_kernel<<<dim3(16, 16, 1), 256, 0, stream>>>(Wq, wqT, D_, D_);
  transpose_cvt_kernel<<<dim3(16, 16, 1), 256, 0, stream>>>(Wk, wkT, D_, D_);
  transpose_cvt_kernel<<<dim3(16, 16, 1), 256, 0, stream>>>(Wo, woT, D_, D_);

  // Q and K projections (each pre-scaled by dh^-0.25)
  proj_kernel<<<dim3(MT_Q, 4), 256, 0, stream>>>(xbf, wqT, qb, MQ, NQ_, 80, scale);
  proj_kernel<<<dim3(MK / 16, 4), 256, 0, stream>>>(ctxb, wkT, kb, MK, NK_, NK_, scale);

  // attention + weighted context
  attn_wctx_kernel<<<dim3(B_, H_), 256, 0, stream>>>(qb, kb, ctxT, wctx);

  // heavy fused Wv contraction (split-K over d, async LDS staging, atomics)
  heavy_kernel<<<dim3(NSLAB, H_), 256, 21504, stream>>>(Wv, xbf, wctx, opre);

  // final output projection
  final_kernel<<<dim3(MT_Q, 4), 256, 0, stream>>>(opre, woT, out);
}
